// StateEliminationNNet_45311904972848
// MI455X (gfx1250) — compile-verified, matrix operand order checked by reading, not run on hardware
//
#include <hip/hip_runtime.h>
#include <hip/hip_bf16.h>

typedef __attribute__((ext_vector_type(16))) _Float16 v16h;
typedef __attribute__((ext_vector_type(8)))  _Float16 v8h;
typedef __attribute__((ext_vector_type(8)))  float    v8f;

#define N_NODES  12800
#define N_EDGES  102400
#define NGRAPH   256
#define ACTION   50
#define SDW      53
#define XDIMW    55
#define CHN      256
#define FEATP    117     // 53 + 64
#define KPAD1    320     // 289 padded to mult of 32

// ---------------- device helpers ----------------

// branch-free fast transcendentals on native gfx1250 TRANS units
__device__ __forceinline__ float fexp2(float x) { return __builtin_amdgcn_exp2f(x); }
__device__ __forceinline__ float frcp(float x)  { return __builtin_amdgcn_rcpf(x); }

__device__ __forceinline__ float sigf(float x) {
  // 1/(1+exp(-x)) = rcp(1 + exp2(-x*log2e))
  return frcp(1.f + fexp2(-1.44269504f * x));
}

__device__ __forceinline__ float ftanh(float x) {
#if __has_builtin(__builtin_amdgcn_tanhf)
  return __builtin_amdgcn_tanhf(x);          // v_tanh_f32
#else
  // tanh(x) = 1 - 2/(exp2(2*log2e*x)+1)
  return 1.f - 2.f * frcp(fexp2(2.88539008f * x) + 1.f);
#endif
}

// Load a 16x32 f16 fragment (A layout; B layout is the mirror with r = N index)
// from a tile stored as 16 rows x 32 halfs, row stride 32 halfs.
// lanes 0-15: row=lane, K 0-7 -> halves 0..7, K 16-23 -> halves 8..15
// lanes 16-31: row=lane-16, K 8-15 and K 24-31.
__device__ __forceinline__ v16h frag_ld(const _Float16* tile, int lane) {
  const int r  = lane & 15;
  const int kh = (lane >> 4) << 3;     // 0 or 8
  const _Float16* p = tile + r * 32;
  v8h lo = *(const v8h*)(p + kh);
  v8h hi = *(const v8h*)(p + 16 + kh);
  v16h o;
#pragma unroll
  for (int i = 0; i < 8; ++i) { o[i] = lo[i]; o[i + 8] = hi[i]; }
  return o;
}

__device__ __forceinline__ void atomicMaxF32(float* a, float v) {
  if (v >= 0.f) atomicMax((int*)a, __float_as_int(v));
  else          atomicMin((unsigned int*)a, (unsigned int)__float_as_int(v));
}

// ---------------- generic small kernels ----------------

__global__ void k_fillf(float* p, int n, float v) {
  int i = blockIdx.x * blockDim.x + threadIdx.x;
  if (i < n) p[i] = v;
}

// pack W [K,N] f32 (row-major) -> Wt [N, Kpad] f16 (transposed, zero padded)
__global__ void k_pack_wt(const float* W, _Float16* Wt, int K, int N, int Kpad) {
  int i = blockIdx.x * blockDim.x + threadIdx.x;
  if (i >= N * Kpad) return;
  int n = i / Kpad, k = i - n * Kpad;
  Wt[i] = (_Float16)(k < K ? W[k * N + n] : 0.f);
}

// C[m,n] = act( b[n] + sum_k A[m,k] * W[k,n] ); act: 0=none 1=tanh 2=elu
__global__ void k_dense(const float* A, const float* W, const float* b, float* C,
                        int M, int K, int N, int act) {
  int i = blockIdx.x * blockDim.x + threadIdx.x;
  if (i >= M * N) return;
  int m = i / N, n = i - m * N;
  float s = b[n];
  for (int k = 0; k < K; ++k) s += A[m * K + k] * W[k * N + n];
  if (act == 1) s = ftanh(s);
  else if (act == 2) s = s > 0.f ? s : expm1f(s);
  C[i] = s;
}

// ---------------- LSTM ----------------

// tabF[v][g] = emb[v]@WihF.T[g] + bihF[g] + bhhF[g]; same for tabB; WhhH = f16(WhhF)
__global__ void k_prep_lstm(const float* emb,
                            const float* WihF, const float* bihF, const float* bhhF,
                            const float* WihB, const float* bihB, const float* bhhB,
                            const float* WhhF,
                            float* tabF, float* tabB, _Float16* WhhH) {
  int i = blockIdx.x * blockDim.x + threadIdx.x;
  if (i < 64 * 128) {
    int v = i >> 7, g = i & 127;
    float sf = bihF[g] + bhhF[g], sb = bihB[g] + bhhB[g];
    for (int k = 0; k < 32; ++k) {
      float e = emb[v * 32 + k];
      sf += e * WihF[g * 32 + k];
      sb += e * WihB[g * 32 + k];
    }
    tabF[i] = sf; tabB[i] = sb;
  }
  if (i < 128 * 32) WhhH[i] = (_Float16)WhhF[i];
}

// Forward LSTM: wave handles 16 edges, block (8 waves) handles 128 edges.
// gates = tabF[tok] + h @ Whh.T  via v_wmma_f32_16x16x32_f16 (8 gate tiles).
__global__ void __launch_bounds__(256)
k_lstm_fwd(const int* tokens, const float* tabF_g, const _Float16* WhhH,
           float* regexF32, _Float16* regexH) {
  __shared__ float tabS[64 * 128];                       // 32 KB
  __shared__ __align__(16) _Float16 hbuf[8][16 * 32];    // 8 KB
  const int tid = threadIdx.x, lane = tid & 31, wave = tid >> 5;
  for (int i = tid; i < 64 * 128; i += 256) tabS[i] = tabF_g[i];

  const int ebase = blockIdx.x * 128 + wave * 16;
  v16h bfrag[8];                                         // Whh^T B fragments
#pragma unroll
  for (int nt = 0; nt < 8; ++nt) bfrag[nt] = frag_ld(WhhH + nt * 16 * 32, lane);

  v16h hfrag;
#pragma unroll
  for (int i = 0; i < 16; ++i) hfrag[i] = (_Float16)0.f;
  v8f c0, c1;
#pragma unroll
  for (int i = 0; i < 8; ++i) { c0[i] = 0.f; c1[i] = 0.f; }

  const int mrow = 8 * (lane >> 4);
  const int ncol = lane & 15;
  __syncthreads();

  for (int t = 0; t < 50; ++t) {
    int tok[8];
#pragma unroll
    for (int r = 0; r < 8; ++r) tok[r] = tokens[(ebase + r + mrow) * 50 + t];
    if (t + 1 < 50)
      __builtin_prefetch(&tokens[(ebase + mrow) * 50 + t + 1], 0, 0);
    v8f acc[8];
#pragma unroll
    for (int nt = 0; nt < 8; ++nt)
#pragma unroll
      for (int r = 0; r < 8; ++r)
        acc[nt][r] = tabS[tok[r] * 128 + nt * 16 + ncol];
#pragma unroll
    for (int nt = 0; nt < 8; ++nt)
      acc[nt] = __builtin_amdgcn_wmma_f32_16x16x32_f16(
          false, hfrag, false, bfrag[nt], (short)0, acc[nt], false, false);
    __syncthreads();
    _Float16* hb = hbuf[wave];
#pragma unroll
    for (int ut = 0; ut < 2; ++ut) {
      v8f* cc = ut ? &c1 : &c0;
#pragma unroll
      for (int r = 0; r < 8; ++r) {
        float iv = sigf(acc[0 + ut][r]);
        float fv = sigf(acc[2 + ut][r]);
        float gv = ftanh(acc[4 + ut][r]);
        float ov = sigf(acc[6 + ut][r]);
        float cn = fv * (*cc)[r] + iv * gv;
        (*cc)[r] = cn;
        hb[(r + mrow) * 32 + ut * 16 + ncol] = (_Float16)(ov * ftanh(cn));
      }
    }
    __syncthreads();
    hfrag = frag_ld(hbuf[wave], lane);
  }
  // write hf -> regex[:, 0:32]
  const _Float16* hb = hbuf[wave];
  for (int j = lane; j < 512; j += 32) {
    int el = j >> 5, u = j & 31;
    int e = ebase + el;
    regexF32[e * 64 + u] = (float)hb[j];
    regexH[e * 64 + u] = hb[j];
  }
}

// Backward direction = single cell step with zero state: pure table + elementwise
__global__ void k_lstm_bwd(const int* tokens, const float* tabB,
                           float* regexF32, _Float16* regexH) {
  int e = blockIdx.x * blockDim.x + threadIdx.x;
  if (e >= N_EDGES) return;
  const float* g = tabB + tokens[e * 50 + 49] * 128;
  for (int u = 0; u < 32; ++u) {
    float cn = sigf(g[u]) * ftanh(g[64 + u]);
    float hv = sigf(g[96 + u]) * ftanh(cn);
    regexF32[e * 64 + 32 + u] = hv;
    regexH[e * 64 + 32 + u] = (_Float16)hv;
  }
}

// ---------------- mean-pool scatter + node feature assembly ----------------

__global__ void k_scatter_cnt(const int* ei, float* cntS, float* cntD) {
  int e = blockIdx.x * blockDim.x + threadIdx.x;
  if (e >= N_EDGES) return;
  atomicAdd(&cntS[ei[e]], 1.f);
  atomicAdd(&cntD[ei[N_EDGES + e]], 1.f);
}

__global__ void k_scatter_feat(const int* ei, const float* srcsn, const float* tgtsn,
                               const float* regexF32, float* sumIn, float* sumOut) {
  int i = blockIdx.x * blockDim.x + threadIdx.x;
  if (i >= N_EDGES * FEATP) return;
  int e = i / FEATP, f = i - e * FEATP;
  float rg = (f >= SDW) ? regexF32[e * 64 + (f - SDW)] : 0.f;
  float vo = (f < SDW) ? tgtsn[e * SDW + f] : rg;
  float vi = (f < SDW) ? srcsn[e * SDW + f] : rg;
  atomicAdd(&sumOut[ei[e] * FEATP + f], vo);             // by src
  atomicAdd(&sumIn[ei[N_EDGES + e] * FEATP + f], vi);    // by dst
}

// hH[n, 0:320] = f16 of concat(x[55], in_tr[117], out_tr[117], zeros[31])
__global__ void k_assemble(const float* x, const float* sumIn, const float* sumOut,
                           const float* cntD, const float* cntS, _Float16* hH) {
  int i = blockIdx.x * blockDim.x + threadIdx.x;
  if (i >= N_NODES * KPAD1) return;
  int n = i / KPAD1, c = i - n * KPAD1;
  float v = 0.f;
  if (c < XDIMW)            v = x[n * XDIMW + c];
  else if (c < XDIMW + FEATP)       v = sumIn[n * FEATP + (c - XDIMW)] / fmaxf(cntD[n], 1.f);
  else if (c < XDIMW + 2 * FEATP)   v = sumOut[n * FEATP + (c - XDIMW - FEATP)] / fmaxf(cntS[n], 1.f);
  hH[i] = (_Float16)v;
}

// ---------------- tiled WMMA GEMM: C[M,N] = A[M,K](f16) * Bt[N,K]^T (f16) ----------------
// block = 256 threads (8 waves), tile 128x64; M%128==0, N%64==0, K%32==0.
__global__ void __launch_bounds__(256)
k_gemm_wmma(const _Float16* A, int lda, const _Float16* Bt, int ldb,
            float* C, int ldc, int K) {
  __shared__ __align__(16) _Float16 As[128 * 32];
  __shared__ __align__(16) _Float16 Bs[64 * 32];
  const int tid = threadIdx.x, lane = tid & 31, wave = tid >> 5;
  const int m0 = blockIdx.x * 128, n0 = blockIdx.y * 64;
  v8f acc[4];
#pragma unroll
  for (int j = 0; j < 4; ++j)
#pragma unroll
    for (int r = 0; r < 8; ++r) acc[j][r] = 0.f;

  for (int k0 = 0; k0 < K; k0 += 32) {
    {
      int r = tid >> 1, part = (tid & 1) * 16;
      const _Float16* s = A + (size_t)(m0 + r) * lda + k0 + part;
      *(v8h*)&As[r * 32 + part]     = *(const v8h*)s;
      *(v8h*)&As[r * 32 + part + 8] = *(const v8h*)(s + 8);
      if (k0 + 32 < K) __builtin_prefetch(s + 32, 0, 0);  // next K panel of A
    }
    {
      int r = tid >> 2, part = (tid & 3) * 8;
      const _Float16* s = Bt + (size_t)(n0 + r) * ldb + k0 + part;
      *(v8h*)&Bs[r * 32 + part] = *(const v8h*)s;
      if (k0 + 32 < K) __builtin_prefetch(s + 32, 0, 0);  // next K panel of B
    }
    __syncthreads();
    v16h af = frag_ld(&As[wave * 16 * 32], lane);
#pragma unroll
    for (int j = 0; j < 4; ++j) {
      v16h bf = frag_ld(&Bs[j * 16 * 32], lane);
      acc[j] = __builtin_amdgcn_wmma_f32_16x16x32_f16(
          false, af, false, bf, (short)0, acc[j], false, false);
    }
    __syncthreads();
  }
  const int mr = m0 + wave * 16 + 8 * (lane >> 4);
  const int nc = n0 + (lane & 15);
#pragma unroll
  for (int j = 0; j < 4; ++j)
#pragma unroll
    for (int r = 0; r < 8; ++r)
      C[(size_t)(mr + r) * ldc + nc + j * 16] = acc[j][r];
}

// ---------------- GAT attention ----------------

__global__ void k_score(const int* ei, const float* xl, const float* xr,
                        const float* ea, const float* att,
                        float* escore, float* mMax) {
  int i = blockIdx.x * blockDim.x + threadIdx.x;
  if (i >= N_EDGES * 8) return;
  int e = i >> 3, h = i & 7;
  int s = ei[e], d = ei[N_EDGES + e];
  const float* pa = att + h * 32;
  float sc = 0.f;
  for (int c = 0; c < 32; ++c) {
    float z = xl[s * CHN + h * 32 + c] + xr[d * CHN + h * 32 + c] + ea[e * CHN + h * 32 + c];
    z = z > 0.f ? z : 0.2f * z;
    sc += z * pa[c];
  }
  escore[i] = sc;
  atomicMaxF32(&mMax[d * 8 + h], sc);
}

__global__ void k_denom(const int* ei, const float* escore, const float* mMax,
                        float* denom) {
  int i = blockIdx.x * blockDim.x + threadIdx.x;
  if (i >= N_EDGES * 8) return;
  int e = i >> 3, h = i & 7;
  int d = ei[N_EDGES + e];
  atomicAdd(&denom[d * 8 + h], __expf(escore[i] - mMax[d * 8 + h]));
}

__global__ void k_biasinit(float* hout, const float* bias) {
  int i = blockIdx.x * blockDim.x + threadIdx.x;
  if (i >= N_NODES * CHN) return;
  hout[i] = bias[i & 255];
}

__global__ void k_aggregate(const int* ei, const float* xl, const float* escore,
                            const float* mMax, const float* denom, float* hout) {
  int i = blockIdx.x * blockDim.x + threadIdx.x;
  if (i >= N_EDGES * CHN) return;
  int e = i >> 8, ch = i & 255, h = ch >> 5;
  int s = ei[e], d = ei[N_EDGES + e];
  float al = __expf(escore[e * 8 + h] - mMax[d * 8 + h]) / denom[d * 8 + h];
  atomicAdd(&hout[d * CHN + ch], al * xl[s * CHN + ch]);
}

__global__ void k_elu_cast(float* hout, _Float16* hH) {
  int i = blockIdx.x * blockDim.x + threadIdx.x;
  if (i >= N_NODES * CHN) return;
  int n = i >> 8, c = i & 255;
  float v = hout[i];
  v = v > 0.f ? v : expm1f(v);
  hout[i] = v;
  hH[n * KPAD1 + c] = (_Float16)v;
}

// ---------------- heads ----------------

__global__ void k_pool(const float* hout, float* sBuf) {
  int i = blockIdx.x * blockDim.x + threadIdx.x;
  if (i >= NGRAPH * CHN) return;
  int g = i >> 8, c = i & 255;
  float s = 0.f;
  for (int a = 0; a < ACTION; ++a) s += hout[(g * ACTION + a) * CHN + c];
  sBuf[i] = s * (1.f / ACTION);
}

__global__ void k_bias_tanh(float* p, const float* b, int M, int N) {
  int i = blockIdx.x * blockDim.x + threadIdx.x;
  if (i >= M * N) return;
  p[i] = ftanh(p[i] + b[i % N]);
}

__global__ void k_logsoftmax(const float* logits, float* out) {
  __shared__ float red[64];
  const int g = blockIdx.x, tid = threadIdx.x;
  float mx = -1e30f;
  for (int a = tid; a < ACTION; a += 64) mx = fmaxf(mx, logits[g * ACTION + a]);
  red[tid] = mx; __syncthreads();
  for (int s = 32; s > 0; s >>= 1) { if (tid < s) red[tid] = fmaxf(red[tid], red[tid + s]); __syncthreads(); }
  float M = red[0]; __syncthreads();
  float sm = 0.f;
  for (int a = tid; a < ACTION; a += 64) sm += __expf(logits[g * ACTION + a] - M);
  red[tid] = sm; __syncthreads();
  for (int s = 32; s > 0; s >>= 1) { if (tid < s) red[tid] += red[tid + s]; __syncthreads(); }
  float L = logf(red[0]);
  for (int a = tid; a < ACTION; a += 64)
    out[g * ACTION + a] = logits[g * ACTION + a] - M - L;
}

// ---------------- launch ----------------

extern "C" void kernel_launch(void* const* d_in, const int* in_sizes, int n_in,
                              void* d_out, int out_size, void* d_ws, size_t ws_size,
                              hipStream_t stream) {
  (void)in_sizes; (void)n_in; (void)out_size; (void)ws_size;
  const float* x     = (const float*)d_in[0];
  const int*   ei    = (const int*)d_in[1];
  const int*   toks  = (const int*)d_in[3];
  const float* srcsn = (const float*)d_in[4];
  const float* tgtsn = (const float*)d_in[5];
  const float* emb   = (const float*)d_in[6];
  const float* WihF  = (const float*)d_in[7];
  const float* WhhF  = (const float*)d_in[8];
  const float* bihF  = (const float*)d_in[9];
  const float* bhhF  = (const float*)d_in[10];
  const float* WihB  = (const float*)d_in[11];
  const float* bihB  = (const float*)d_in[13];
  const float* bhhB  = (const float*)d_in[14];
  const float* Wl[3]  = {(const float*)d_in[15], (const float*)d_in[20], (const float*)d_in[25]};
  const float* Wr[3]  = {(const float*)d_in[16], (const float*)d_in[21], (const float*)d_in[26]};
  const float* We[3]  = {(const float*)d_in[17], (const float*)d_in[22], (const float*)d_in[27]};
  const float* attp[3]= {(const float*)d_in[18], (const float*)d_in[23], (const float*)d_in[28]};
  const float* bp[3]  = {(const float*)d_in[19], (const float*)d_in[24], (const float*)d_in[29]};
  const float* polW[4] = {(const float*)d_in[30], (const float*)d_in[32], (const float*)d_in[34], (const float*)d_in[36]};
  const float* polB[4] = {(const float*)d_in[31], (const float*)d_in[33], (const float*)d_in[35], (const float*)d_in[37]};
  const float* valW[4] = {(const float*)d_in[38], (const float*)d_in[40], (const float*)d_in[42], (const float*)d_in[44]};
  const float* valB[4] = {(const float*)d_in[39], (const float*)d_in[41], (const float*)d_in[43], (const float*)d_in[45]};
  float* out = (float*)d_out;

  char* wsp = (char*)d_ws;
  size_t off = 0;
  auto alloc = [&](size_t bytes) -> void* {
    void* p = wsp + off;
    off += (bytes + 255) & ~(size_t)255;
    return p;
  };
  float*     regexF32 = (float*)alloc((size_t)N_EDGES * 64 * 4);
  _Float16*  regexH   = (_Float16*)alloc((size_t)N_EDGES * 64 * 2);
  float*     tabF     = (float*)alloc(64 * 128 * 4);
  float*     tabB     = (float*)alloc(64 * 128 * 4);
  _Float16*  WhhH     = (_Float16*)alloc(128 * 32 * 2);
  _Float16*  hH       = (_Float16*)alloc((size_t)N_NODES * KPAD1 * 2);
  float*     sumIn    = (float*)alloc((size_t)N_NODES * FEATP * 4);
  float*     sumOut   = (float*)alloc((size_t)N_NODES * FEATP * 4);
  float*     cntS     = (float*)alloc(N_NODES * 4);
  float*     cntD     = (float*)alloc(N_NODES * 4);
  _Float16*  wtA      = (_Float16*)alloc(256 * KPAD1 * 2);
  _Float16*  wtB      = (_Float16*)alloc(256 * KPAD1 * 2);
  _Float16*  wtE      = (_Float16*)alloc(256 * 64 * 2);
  _Float16*  wtP      = (_Float16*)alloc(128 * 256 * 2);
  float*     xl       = (float*)alloc((size_t)N_NODES * CHN * 4);
  float*     xr       = (float*)alloc((size_t)N_NODES * CHN * 4);
  float*     ea       = (float*)alloc((size_t)N_EDGES * CHN * 4);
  float*     escore   = (float*)alloc((size_t)N_EDGES * 8 * 4);
  float*     mMax     = (float*)alloc(N_NODES * 8 * 4);
  float*     denom    = (float*)alloc(N_NODES * 8 * 4);
  float*     hout     = (float*)alloc((size_t)N_NODES * CHN * 4);
  float*     sBuf     = (float*)alloc(NGRAPH * CHN * 4);
  float*     p1       = (float*)alloc((size_t)N_NODES * 128 * 4);
  float*     p2       = (float*)alloc((size_t)N_NODES * 64 * 4);
  float*     p3       = (float*)alloc((size_t)N_NODES * 32 * 4);
  float*     logits   = (float*)alloc(N_NODES * 4);
  float*     v1       = (float*)alloc(NGRAPH * 128 * 4);
  float*     v2       = (float*)alloc(NGRAPH * 64 * 4);
  float*     v3       = (float*)alloc(NGRAPH * 32 * 4);

  auto cdiv = [](int a, int b) { return (a + b - 1) / b; };

  // ---- regex bi-LSTM ----
  k_prep_lstm<<<32, 256, 0, stream>>>(emb, WihF, bihF, bhhF, WihB, bihB, bhhB,
                                      WhhF, tabF, tabB, WhhH);
  k_lstm_fwd<<<N_EDGES / 128, 256, 0, stream>>>(toks, tabF, WhhH, regexF32, regexH);
  k_lstm_bwd<<<N_EDGES / 256, 256, 0, stream>>>(toks, tabB, regexF32, regexH);

  // ---- transition mean pools + node feature assembly ----
  k_fillf<<<cdiv(N_NODES * FEATP, 256), 256, 0, stream>>>(sumIn, N_NODES * FEATP, 0.f);
  k_fillf<<<cdiv(N_NODES * FEATP, 256), 256, 0, stream>>>(sumOut, N_NODES * FEATP, 0.f);
  k_fillf<<<cdiv(N_NODES, 256), 256, 0, stream>>>(cntS, N_NODES, 0.f);
  k_fillf<<<cdiv(N_NODES, 256), 256, 0, stream>>>(cntD, N_NODES, 0.f);
  k_scatter_cnt<<<N_EDGES / 256, 256, 0, stream>>>(ei, cntS, cntD);
  k_scatter_feat<<<cdiv(N_EDGES * FEATP, 256), 256, 0, stream>>>(ei, srcsn, tgtsn,
                                                                 regexF32, sumIn, sumOut);
  k_assemble<<<cdiv(N_NODES * KPAD1, 256), 256, 0, stream>>>(x, sumIn, sumOut, cntD, cntS, hH);

  // ---- 3 GAT convolutions ----
  for (int cv = 0; cv < 3; ++cv) {
    const int K = (cv == 0) ? 289 : 256;
    const int Kp = (cv == 0) ? KPAD1 : 256;
    k_pack_wt<<<cdiv(256 * Kp, 256), 256, 0, stream>>>(Wl[cv], wtA, K, 256, Kp);
    k_pack_wt<<<cdiv(256 * Kp, 256), 256, 0, stream>>>(Wr[cv], wtB, K, 256, Kp);
    k_pack_wt<<<cdiv(256 * 64, 256), 256, 0, stream>>>(We[cv], wtE, 64, 256, 64);
    k_gemm_wmma<<<dim3(N_NODES / 128, CHN / 64), 256, 0, stream>>>(hH, KPAD1, wtA, Kp, xl, CHN, Kp);
    k_gemm_wmma<<<dim3(N_NODES / 128, CHN / 64), 256, 0, stream>>>(hH, KPAD1, wtB, Kp, xr, CHN, Kp);
    k_gemm_wmma<<<dim3(N_EDGES / 128, CHN / 64), 256, 0, stream>>>(regexH, 64, wtE, 64, ea, CHN, 64);
    k_fillf<<<cdiv(N_NODES * 8, 256), 256, 0, stream>>>(mMax, N_NODES * 8, -1e30f);
    k_fillf<<<cdiv(N_NODES * 8, 256), 256, 0, stream>>>(denom, N_NODES * 8, 0.f);
    k_score<<<cdiv(N_EDGES * 8, 256), 256, 0, stream>>>(ei, xl, xr, ea, attp[cv], escore, mMax);
    k_denom<<<cdiv(N_EDGES * 8, 256), 256, 0, stream>>>(ei, escore, mMax, denom);
    k_biasinit<<<cdiv(N_NODES * CHN, 256), 256, 0, stream>>>(hout, bp[cv]);
    k_aggregate<<<cdiv(N_EDGES * CHN, 256), 256, 0, stream>>>(ei, xl, escore, mMax, denom, hout);
    k_elu_cast<<<cdiv(N_NODES * CHN, 256), 256, 0, stream>>>(hout, hH);
  }

  // ---- value head ----
  k_pool<<<cdiv(NGRAPH * CHN, 256), 256, 0, stream>>>(hout, sBuf);
  k_dense<<<cdiv(NGRAPH * 128, 256), 256, 0, stream>>>(sBuf, valW[0], valB[0], v1, NGRAPH, 256, 128, 2);
  k_dense<<<cdiv(NGRAPH * 64, 256), 256, 0, stream>>>(v1, valW[1], valB[1], v2, NGRAPH, 128, 64, 2);
  k_dense<<<cdiv(NGRAPH * 32, 256), 256, 0, stream>>>(v2, valW[2], valB[2], v3, NGRAPH, 64, 32, 2);
  k_dense<<<cdiv(NGRAPH, 256), 256, 0, stream>>>(v3, valW[3], valB[3], out + N_NODES, NGRAPH, 32, 1, 0);

  // ---- policy head ----
  k_pack_wt<<<cdiv(128 * 256, 256), 256, 0, stream>>>(polW[0], wtP, 256, 128, 256);
  k_gemm_wmma<<<dim3(N_NODES / 128, 128 / 64), 256, 0, stream>>>(hH, KPAD1, wtP, 256, p1, 128, 256);
  k_bias_tanh<<<cdiv(N_NODES * 128, 256), 256, 0, stream>>>(p1, polB[0], N_NODES, 128);
  k_dense<<<cdiv(N_NODES * 64, 256), 256, 0, stream>>>(p1, polW[1], polB[1], p2, N_NODES, 128, 64, 1);
  k_dense<<<cdiv(N_NODES * 32, 256), 256, 0, stream>>>(p2, polW[2], polB[2], p3, N_NODES, 64, 32, 1);
  k_dense<<<cdiv(N_NODES, 256), 256, 0, stream>>>(p3, polW[3], polB[3], logits, N_NODES, 32, 1, 0);
  k_logsoftmax<<<NGRAPH, 64, 0, stream>>>(logits, out);
}
